// CompositionalFC_88527865905610
// MI455X (gfx1250) — compile-verified, matrix operand order checked by reading, not run on hardware
//
#include <hip/hip_runtime.h>
#include <stdint.h>

#define NE    16
#define BATCH 4096
#define KDIM  1024
#define ODIM  1024

#define BM 128
#define BN 128
#define BK 64

typedef __attribute__((ext_vector_type(16))) __bf16 v16bf;
typedef __attribute__((ext_vector_type(8)))  __bf16 v8bf;
typedef __attribute__((ext_vector_type(8)))  float  v8f;

#if defined(__has_builtin)
#if __has_builtin(__builtin_amdgcn_global_load_async_to_lds_b128)
#define HAVE_ASYNC_LDS 1
#endif
#if __has_builtin(__builtin_amdgcn_s_wait_asynccnt)
#define HAVE_WAIT_ASYNC 1
#endif
#endif

typedef int v4i_t __attribute__((vector_size(16)));

__device__ __forceinline__ void async_copy16(const void* g, void* l) {
#ifdef HAVE_ASYNC_LDS
  __builtin_amdgcn_global_load_async_to_lds_b128(
      (__attribute__((address_space(1))) v4i_t*)(void*)g,
      (__attribute__((address_space(3))) v4i_t*)l,
      0, 0);
#else
  *(uint4*)l = *(const uint4*)g;
#endif
}

__device__ __forceinline__ void wait_async0() {
#ifdef HAVE_ASYNC_LDS
#ifdef HAVE_WAIT_ASYNC
  __builtin_amdgcn_s_wait_asynccnt(0);
#else
  asm volatile("s_wait_asynccnt 0" ::: "memory");
#endif
#endif
}

__device__ __forceinline__ v8f splat8(float s) {
  v8f v;
#pragma unroll
  for (int i = 0; i < 8; ++i) v[i] = s;
  return v;
}

// ---------------- fp32 -> bf16 convert for x ----------------
__global__ __launch_bounds__(256) void cvt_x_kernel(const float* __restrict__ x,
                                                    unsigned short* __restrict__ xb) {
  size_t i = ((size_t)blockIdx.x * blockDim.x + threadIdx.x) * 4;
  float4 v = *(const float4*)&x[i];
  __attribute__((aligned(8))) __bf16 t[4] = {(__bf16)v.x, (__bf16)v.y, (__bf16)v.z, (__bf16)v.w};
  *(uint2*)&xb[i] = *(const uint2*)t;
}

// ------- fp32 -> bf16 convert + transpose W[n][i][o] -> Wt[n][o][i] -------
__global__ __launch_bounds__(256) void cvt_wt_kernel(const float* __restrict__ w,
                                                     unsigned short* __restrict__ wtb) {
  __shared__ float tile[32][33];
  const int n  = blockIdx.z;
  const int ib = blockIdx.y * 32;
  const int ob = blockIdx.x * 32;
  const int r  = threadIdx.x >> 3;
  const int c4 = (threadIdx.x & 7) * 4;

  float4 v = *(const float4*)&w[((size_t)n * KDIM + ib + r) * ODIM + ob + c4];
  tile[r][c4 + 0] = v.x; tile[r][c4 + 1] = v.y;
  tile[r][c4 + 2] = v.z; tile[r][c4 + 3] = v.w;
  __syncthreads();

  __attribute__((aligned(8))) __bf16 t[4];
#pragma unroll
  for (int j = 0; j < 4; ++j) t[j] = (__bf16)tile[c4 + j][r];
  *(uint2*)&wtb[((size_t)n * ODIM + ob + r) * KDIM + ib + c4] = *(const uint2*)t;
}

// ---------------- fused 16-expert WMMA GEMM + combine ----------------
__global__ __launch_bounds__(256) void moe_wmma_kernel(
    const unsigned short* __restrict__ xb,   // [BATCH][KDIM]  bf16
    const unsigned short* __restrict__ wtb,  // [NE][ODIM][KDIM] bf16 (K-contig)
    const float* __restrict__ cw,            // [BATCH][NE]
    const float* __restrict__ bias,          // [NE][ODIM]
    float* __restrict__ out)                 // [BATCH][ODIM]
{
  __shared__ __attribute__((aligned(16))) __bf16 As[2][BM * BK];
  __shared__ __attribute__((aligned(16))) __bf16 Bs[2][BN * BK];
  __shared__ float cws[BM * NE];
  __shared__ float biass[NE * BN];

  const int tid   = threadIdx.x;
  const int lane  = tid & 31;
  const int wid   = tid >> 5;
  const int waveM = wid & 1;   // 0..1 -> 64-row slab
  const int waveN = wid >> 1;  // 0..3 -> 32-col slab
  const int bbase = blockIdx.y * BM;
  const int obase = blockIdx.x * BN;
  const int m16   = lane & 15;
  const int hl    = lane >> 4;

  // stage comp_weight and bias slices once
  for (int idx = tid; idx < BM * NE; idx += 256) {
    int r = idx >> 4, e = idx & 15;
    cws[idx] = cw[(size_t)(bbase + r) * NE + e];
  }
  for (int idx = tid; idx < NE * BN; idx += 256) {
    int e = idx >> 7, c = idx & 127;
    biass[idx] = bias[(size_t)e * ODIM + obase + c];
  }
  __syncthreads();

  v8f acc[4][2];
  v8f z[4][2];
#pragma unroll
  for (int im = 0; im < 4; ++im)
#pragma unroll
    for (int in2 = 0; in2 < 2; ++in2) {
      acc[im][in2] = splat8(0.f);
      z[im][in2]   = splat8(biass[0 * BN + waveN * 32 + in2 * 16 + m16]);
    }

  auto issue_stage = [&](int st, int buf) {
    const int n1 = st >> 4;
    const int kb = (st & 15) * BK;
#pragma unroll
    for (int j = 0; j < 4; ++j) {
      int c   = tid + 256 * j;       // 0..1023 16-byte chunks
      int row = c >> 3;              // 8 chunks per 128-byte row
      int col = (c & 7) * 8;         // bf16 elements
      async_copy16(&xb[(size_t)(bbase + row) * KDIM + kb + col],
                   &As[buf][row * BK + col]);
      async_copy16(&wtb[((size_t)n1 * ODIM + obase + row) * KDIM + kb + col],
                   &Bs[buf][row * BK + col]);
    }
  };

  issue_stage(0, 0);

  for (int st = 0; st < NE * (KDIM / BK); ++st) {
    const int buf = st & 1;
    wait_async0();
    __syncthreads();
    if (st + 1 < NE * (KDIM / BK)) issue_stage(st + 1, buf ^ 1);

    // two K=32 WMMA steps from this BK=64 stage
#pragma unroll
    for (int ks = 0; ks < 2; ++ks) {
      const int k0 = ks * 32;
      union Frag { v16bf v; v8bf h[2]; };
      Frag afr[4], bfr[2];
#pragma unroll
      for (int im = 0; im < 4; ++im) {
        const __bf16* base = &As[buf][(waveM * 64 + im * 16 + m16) * BK];
        afr[im].h[0] = *(const v8bf*)&base[k0 + hl * 8];
        afr[im].h[1] = *(const v8bf*)&base[k0 + 16 + hl * 8];
      }
#pragma unroll
      for (int in2 = 0; in2 < 2; ++in2) {
        const __bf16* base = &Bs[buf][(waveN * 32 + in2 * 16 + m16) * BK];
        bfr[in2].h[0] = *(const v8bf*)&base[k0 + hl * 8];
        bfr[in2].h[1] = *(const v8bf*)&base[k0 + 16 + hl * 8];
      }
#pragma unroll
      for (int im = 0; im < 4; ++im)
#pragma unroll
        for (int in2 = 0; in2 < 2; ++in2)
          z[im][in2] = __builtin_amdgcn_wmma_f32_16x16x32_bf16(
              false, afr[im].v, false, bfr[in2].v,
              (short)0, z[im][in2], false, false);
    }

    // expert boundary: acc += cw[b][n] * z ; re-init z with next expert's bias
    if ((st & 15) == 15) {
      const int n1 = st >> 4;
#pragma unroll
      for (int im = 0; im < 4; ++im) {
#pragma unroll
        for (int r = 0; r < 8; ++r) {
          float c = cws[(waveM * 64 + im * 16 + r + 8 * hl) * NE + n1];
          acc[im][0][r] += c * z[im][0][r];
          acc[im][1][r] += c * z[im][1][r];
        }
      }
      if (n1 < NE - 1) {
#pragma unroll
        for (int in2 = 0; in2 < 2; ++in2) {
          float b0 = biass[(n1 + 1) * BN + waveN * 32 + in2 * 16 + m16];
#pragma unroll
          for (int im = 0; im < 4; ++im)
#pragma unroll
            for (int r = 0; r < 8; ++r) z[im][in2][r] = b0;
        }
      }
    }
  }

  // ReLU + store
#pragma unroll
  for (int im = 0; im < 4; ++im) {
#pragma unroll
    for (int in2 = 0; in2 < 2; ++in2) {
      const int go = obase + waveN * 32 + in2 * 16 + m16;
#pragma unroll
      for (int r = 0; r < 8; ++r) {
        const int gb = bbase + waveM * 64 + im * 16 + r + 8 * hl;
        float v = acc[im][in2][r];
        out[(size_t)gb * ODIM + go] = v > 0.f ? v : 0.f;
      }
    }
  }
}

extern "C" void kernel_launch(void* const* d_in, const int* in_sizes, int n_in,
                              void* d_out, int out_size, void* d_ws, size_t ws_size,
                              hipStream_t stream) {
  const float* x    = (const float*)d_in[0];
  const float* cw   = (const float*)d_in[1];
  const float* w    = (const float*)d_in[2];
  const float* bias = (const float*)d_in[3];
  float* out = (float*)d_out;

  unsigned short* xb  = (unsigned short*)d_ws;               // 4096*1024 bf16 (8 MB)
  unsigned short* wtb = xb + (size_t)BATCH * KDIM;           // 16*1024*1024 bf16 (32 MB)

  cvt_x_kernel<<<(BATCH * KDIM) / (256 * 4), 256, 0, stream>>>(x, xb);
  cvt_wt_kernel<<<dim3(ODIM / 32, KDIM / 32, NE), 256, 0, stream>>>(w, wtb);
  moe_wmma_kernel<<<dim3(ODIM / BN, BATCH / BM), 256, 0, stream>>>(xb, wtb, cw, bias, out);
}
